// CIN0_867583394519
// MI455X (gfx1250) — matrix operand order repaired and checked by reading, not compiled, not run on hardware
//
#include <hip/hip_runtime.h>
#include <hip/hip_bf16.h>

// MI455X / gfx1250 implementation of the CIN reference.
// Dense math (message MLPs, update MLPs, head) on v_wmma_f32_16x16x32_f16
// (f16 inputs, f32 accumulation). segment_sum -> global_atomic_add_f32.
// Weights are pre-transposed + f16-converted in workspace. The persistent
// message kernel keeps its B fragments in registers; the persistent update
// kernel stages its weights into LDS once per block via the CDNA5 async
// copy path (global_load_async_to_lds_b128 / s_wait_asynccnt).

typedef __attribute__((ext_vector_type(16))) _Float16 v16h;
typedef __attribute__((ext_vector_type(8)))  _Float16 v8h;
typedef __attribute__((ext_vector_type(8)))  float    v8f;

#define BN_EPS 1e-5f

// ---- WMMA fragment layouts (wave32) ---------------------------------------
// A (16x32 f16): lane m=lane&15 is the row; elements 0..7 hold K=8*half..+7,
// elements 8..15 hold K=16+8*half..+7  => two pairs of contiguous float4.
// B (32x16 f16): lane n=lane&15 is the column; same K order as A
//   => with transposed f16 weights Wt[n*K+k], two b128 loads at 8*half, 16+8*half.
// C/D (16x16 f32, v8f): element r -> (M = r + 8*half, N = lane&15)

__device__ __forceinline__ v8f v8f_zero() {
    v8f z = {0.f,0.f,0.f,0.f,0.f,0.f,0.f,0.f};
    return z;
}

// A fragment for one 32-wide K block starting at p (row-contiguous f32).
__device__ __forceinline__ v16h load_afrag(const float* p, int half) {
    const float4* q0 = (const float4*)(p + (half << 3));
    const float4* q1 = (const float4*)(p + 16 + (half << 3));
    float4 f0 = q0[0], f1 = q0[1], f2 = q1[0], f3 = q1[1];
    v16h r;
    r[0]=(_Float16)f0.x;  r[1]=(_Float16)f0.y;  r[2]=(_Float16)f0.z;  r[3]=(_Float16)f0.w;
    r[4]=(_Float16)f1.x;  r[5]=(_Float16)f1.y;  r[6]=(_Float16)f1.z;  r[7]=(_Float16)f1.w;
    r[8]=(_Float16)f2.x;  r[9]=(_Float16)f2.y;  r[10]=(_Float16)f2.z; r[11]=(_Float16)f2.w;
    r[12]=(_Float16)f3.x; r[13]=(_Float16)f3.y; r[14]=(_Float16)f3.z; r[15]=(_Float16)f3.w;
    return r;
}

// A fragment from f16 data (LDS-staged): two b128 loads.
__device__ __forceinline__ v16h load_afrag16(const _Float16* p, int half) {
    v8h lo = *(const v8h*)(p + (half << 3));
    v8h hi = *(const v8h*)(p + 16 + (half << 3));
    v16h r;
#pragma unroll
    for (int e = 0; e < 8; ++e) { r[e] = lo[e]; r[e + 8] = hi[e]; }
    return r;
}

// B fragment from transposed f16 weights: 8 halfs at p+8*half and p+16+8*half.
__device__ __forceinline__ v16h load_bfrag16(const _Float16* p, int half) {
    const _Float16* q = p + (half << 3);
    v8h lo = *(const v8h*)(q);
    v8h hi = *(const v8h*)(q + 16);
    v16h r;
#pragma unroll
    for (int e = 0; e < 8; ++e) { r[e] = lo[e]; r[e + 8] = hi[e]; }
    return r;
}

// Async copy of `halfs` f16 elements (multiple of 8) from global `g` to LDS
// `l`, distributed over the whole block; must be followed by wait + barrier.
__device__ __forceinline__ void async_stage_weights(const _Float16* g, _Float16* l,
                                                    int halfs) {
    const int chunks = halfs >> 3;                 // 16-byte chunks
    for (int c = threadIdx.x; c < chunks; c += blockDim.x) {
        unsigned lds = (unsigned)(size_t)(l + (c << 3));   // wg-relative LDS offset
        asm volatile("global_load_async_to_lds_b128 %0, %1, %2"
                     :: "v"(lds), "v"(c << 4), "s"(g) : "memory");
    }
    asm volatile("s_wait_asynccnt 0" ::: "memory");
}

// ---- small utility kernels -------------------------------------------------
__global__ void scale2_kernel(const float* __restrict__ x, float* __restrict__ acc,
                              unsigned long long n) {
    unsigned long long i = (unsigned long long)blockIdx.x * blockDim.x + threadIdx.x;
    if (i < n) acc[i] = 2.0f * x[i];
}

__global__ void zero_kernel(float* __restrict__ p, int n) {
    int i = blockIdx.x * blockDim.x + threadIdx.x;
    if (i < n) p[i] = 0.f;
}

// Wt16[n*K + k] = (f16)W[k*N + n], zero-padding columns n in [N, Npad)
__global__ void transpose_w16_kernel(const float* __restrict__ W, _Float16* __restrict__ Wt,
                                     int K, int N, int Npad) {
    int i = blockIdx.x * blockDim.x + threadIdx.x;
    if (i >= K * Npad) return;
    int k = i / Npad, n = i - k * Npad;
    Wt[n * K + k] = (n < N) ? (_Float16)W[k * N + n] : (_Float16)0.f;
}

// ---- message kernel: BN(ReLU(concat(x[src], xo[att]) @ W + b)) scattered ---
// PERSISTENT: grid-stride over 16-edge tiles; B fragments + BN params hoisted
// into registers once per wave. K = 2*LD, Nout = LD.
template<int LD>
__global__ void msg_kernel(const float* __restrict__ x,     // [*, LD]
                           const float* __restrict__ xo,    // [*, LD]
                           const int*   __restrict__ src,   // [E]
                           const int*   __restrict__ dst,   // [E]
                           const int*   __restrict__ att,   // [E]
                           const _Float16* __restrict__ Wt, // [LD, 2*LD] transposed f16
                           const float* __restrict__ bias,
                           const float* __restrict__ gamma,
                           const float* __restrict__ beta,
                           const float* __restrict__ mean,
                           const float* __restrict__ var,
                           float* __restrict__ acc,         // [N, LD] (atomic accum)
                           int E, int stride /* edges per grid sweep */) {
    constexpr int K  = 2 * LD;
    constexpr int NT = LD / 16;   // n tiles
    constexpr int KT = K / 32;    // k steps
    const int lane = threadIdx.x & 31;
    const int wv   = threadIdx.x >> 5;
    const int half = lane >> 4;
    const int m    = lane & 15;
    const int tile0 = (blockIdx.x * (blockDim.x >> 5) + wv) << 4;

    // hoist weight fragments (KT*NT x v16h) and BN params for the whole sweep
    v16h bf[KT][NT];
#pragma unroll
    for (int kk = 0; kk < KT; ++kk)
#pragma unroll
        for (int t = 0; t < NT; ++t)
            bf[kk][t] = load_bfrag16(Wt + (size_t)((t << 4) + m) * K + (kk << 5), half);
    float sc[NT], sh[NT], bi[NT];
#pragma unroll
    for (int t = 0; t < NT; ++t) {
        const int n = (t << 4) + m;
        sc[t] = gamma[n] * rsqrtf(var[n] + BN_EPS);
        sh[t] = beta[n] - mean[n] * sc[t];
        bi[t] = bias[n];
    }

    for (int tile = tile0; tile < E; tile += stride) {   // wave-uniform loop
        const int erow = tile + m;
        const bool valid = erow < E;
        const int s = valid ? src[erow] : 0;    // clamped; garbage rows masked out
        const int a = valid ? att[erow] : 0;    // in the epilogue
        const float* xs = x  + (size_t)s * LD;
        const float* xa = xo + (size_t)a * LD;

        v8f accf[NT];
#pragma unroll
        for (int t = 0; t < NT; ++t) accf[t] = v8f_zero();

#pragma unroll
        for (int kk = 0; kk < KT; ++kk) {
            const int kb = kk << 5;
            // each 32-wide K block comes entirely from one half of the concat
            const float* ap = (kb < LD) ? (xs + kb) : (xa + (kb - LD));
            v16h afrag = load_afrag(ap, half);
#pragma unroll
            for (int t = 0; t < NT; ++t)
                accf[t] = __builtin_amdgcn_wmma_f32_16x16x32_f16(
                    false, afrag, false, bf[kk][t], (short)0, accf[t], false, false);
        }
        // epilogue: +bias, ReLU, BN(eval), atomic scatter-add to acc[dst]
#pragma unroll
        for (int r = 0; r < 8; ++r) {
            const int er = tile + r + (half << 3);
            if (er < E) {
                float* dr = acc + (size_t)dst[er] * LD;
#pragma unroll
                for (int t = 0; t < NT; ++t) {
                    float v = accf[t][r] + bi[t];
                    v = v > 0.f ? v : 0.f;
                    atomicAdd(dr + (t << 4) + m, v * sc[t] + sh[t]);
                }
            }
        }
    }
}

// ---- update kernel: BN(ReLU(ReLU(acc @ W1 + b1) @ W2 + b2)) -> xout [N,64] -
// PERSISTENT: weights async-staged to LDS once per block; grid-stride tiles.
template<int LDIN>
__global__ void update_kernel(const float* __restrict__ xin,     // [N, LDIN]
                              const _Float16* __restrict__ W1t,  // [64, LDIN] f16
                              const float* __restrict__ b1,
                              const _Float16* __restrict__ W2t,  // [64, 64] f16
                              const float* __restrict__ b2,
                              const float* __restrict__ gamma,
                              const float* __restrict__ beta,
                              const float* __restrict__ mean,
                              const float* __restrict__ var,
                              float* __restrict__ xout,          // [N, 64]
                              int N, int stride /* rows per grid sweep */) {
    constexpr int H = 64;
    __shared__ __align__(16) _Float16 w1s[H * LDIN];   // [n][k] transposed
    __shared__ __align__(16) _Float16 w2s[H * H];      // [n][k] transposed
    __shared__ __align__(16) _Float16 hbuf[8][16][H];  // per-wave 16x64 f16 relayout

    // CDNA5 async copy: weights -> LDS, shared by all 8 waves in the block
    async_stage_weights(W1t, w1s, H * LDIN);
    async_stage_weights(W2t, w2s, H * H);
    __syncthreads();

    const int lane = threadIdx.x & 31;
    const int wv   = threadIdx.x >> 5;
    const int half = lane >> 4;
    const int m    = lane & 15;
    const int tile0 = (blockIdx.x * (blockDim.x >> 5) + wv) << 4;

    // hoist biases + BN params for the whole sweep
    float b1v[4], b2v[4], scv[4], shv[4];
#pragma unroll
    for (int t = 0; t < 4; ++t) {
        const int n = (t << 4) + m;
        b1v[t] = b1[n];
        b2v[t] = b2[n];
        scv[t] = gamma[n] * rsqrtf(var[n] + BN_EPS);
        shv[t] = beta[n] - mean[n] * scv[t];
    }

    for (int tile = tile0; tile < N; tile += stride) {   // wave-uniform loop
        const int row = tile + m;
        const float* xr = xin + (size_t)((row < N) ? row : 0) * LDIN;

        // GEMM1: [16 x LDIN] @ [LDIN x 64]
        v8f a1[4];
#pragma unroll
        for (int t = 0; t < 4; ++t) a1[t] = v8f_zero();
#pragma unroll
        for (int kb = 0; kb < LDIN; kb += 32) {
            v16h afrag = load_afrag(xr + kb, half);
#pragma unroll
            for (int t = 0; t < 4; ++t) {
                v16h bfrag = load_bfrag16(&w1s[((t << 4) + m) * LDIN + kb], half);
                a1[t] = __builtin_amdgcn_wmma_f32_16x16x32_f16(
                    false, afrag, false, bfrag, (short)0, a1[t], false, false);
            }
        }
        // h = ReLU(a1 + b1) -> LDS (D-layout -> row-major, f16)
#pragma unroll
        for (int t = 0; t < 4; ++t) {
            const int n = (t << 4) + m;
#pragma unroll
            for (int r = 0; r < 8; ++r) {
                float v = a1[t][r] + b1v[t];
                hbuf[wv][r + (half << 3)][n] = (_Float16)(v > 0.f ? v : 0.f);
            }
        }
        asm volatile("s_wait_dscnt 0" ::: "memory");  // wave-local LDS RAW fence

        // GEMM2: [16 x 64] @ [64 x 64]
        v8f a2[4];
#pragma unroll
        for (int t = 0; t < 4; ++t) a2[t] = v8f_zero();
#pragma unroll
        for (int kb = 0; kb < H; kb += 32) {
            v16h afrag = load_afrag16(&hbuf[wv][m][kb], half);
#pragma unroll
            for (int t = 0; t < 4; ++t) {
                v16h bfrag = load_bfrag16(&w2s[((t << 4) + m) * H + kb], half);
                a2[t] = __builtin_amdgcn_wmma_f32_16x16x32_f16(
                    false, afrag, false, bfrag, (short)0, a2[t], false, false);
            }
        }
        // epilogue: +b2, ReLU, BN -> xout
#pragma unroll
        for (int t = 0; t < 4; ++t) {
            const int n = (t << 4) + m;
#pragma unroll
            for (int r = 0; r < 8; ++r) {
                const int orow = tile + r + (half << 3);
                if (orow < N) {
                    float v = a2[t][r] + b2v[t];
                    v = v > 0.f ? v : 0.f;
                    xout[(size_t)orow * H + n] = v * scv[t] + shv[t];
                }
            }
        }
    }
}

// ---- pooling: pooled[batch[n]] += x[n] ------------------------------------
__global__ void pool_kernel(const float* __restrict__ x, const int* __restrict__ batch,
                            float* __restrict__ pooled, int N) {
    const int i = blockIdx.x * blockDim.x + threadIdx.x;
    if (i >= N * 64) return;
    const int n = i >> 6, c = i & 63;
    atomicAdd(&pooled[batch[n] * 64 + c], x[i]);
}

// ---- head: out = ReLU(pooled @ W1 + b1) @ W2 + b2  (64x64 -> 64x10) -------
__global__ void head_kernel(const float* __restrict__ pooled,   // [64,64]
                            const _Float16* __restrict__ W1t,   // [64,64] f16
                            const float* __restrict__ b1,
                            const _Float16* __restrict__ W2t,   // [16,64] f16, padded
                            const float* __restrict__ b2,
                            float* __restrict__ out) {          // [64,10]
    constexpr int H = 64, C = 10;
    __shared__ __align__(16) _Float16 hbuf[4][16][H];
    const int lane = threadIdx.x & 31;
    const int wv   = threadIdx.x >> 5;
    const int half = lane >> 4;
    const int m    = lane & 15;
    const int tile = wv << 4;

    v8f a1[4];
#pragma unroll
    for (int t = 0; t < 4; ++t) a1[t] = v8f_zero();
#pragma unroll
    for (int kb = 0; kb < H; kb += 32) {
        v16h afrag = load_afrag(pooled + (tile + m) * H + kb, half);
#pragma unroll
        for (int t = 0; t < 4; ++t) {
            const int n = (t << 4) + m;
            v16h bfrag = load_bfrag16(W1t + (size_t)n * H + kb, half);
            a1[t] = __builtin_amdgcn_wmma_f32_16x16x32_f16(
                false, afrag, false, bfrag, (short)0, a1[t], false, false);
        }
    }
#pragma unroll
    for (int t = 0; t < 4; ++t) {
        const int n  = (t << 4) + m;
        const float bi = b1[n];
#pragma unroll
        for (int r = 0; r < 8; ++r) {
            float v = a1[t][r] + bi;
            hbuf[wv][r + (half << 3)][n] = (_Float16)(v > 0.f ? v : 0.f);
        }
    }
    asm volatile("s_wait_dscnt 0" ::: "memory");

    // GEMM2 with N padded 10 -> 16 (zero columns already in W2t)
    v8f a2 = v8f_zero();
#pragma unroll
    for (int kb = 0; kb < H; kb += 32) {
        v16h afrag = load_afrag16(&hbuf[wv][m][kb], half);
        v16h bfrag = load_bfrag16(W2t + (size_t)m * H + kb, half);
        a2 = __builtin_amdgcn_wmma_f32_16x16x32_f16(
            false, afrag, false, bfrag, (short)0, a2, false, false);
    }
    if (m < C) {
        const float bi = b2[m];
#pragma unroll
        for (int r = 0; r < 8; ++r)
            out[(tile + r + (half << 3)) * C + m] = a2[r] + bi;
    }
}

// ---------------------------------------------------------------------------
extern "C" void kernel_launch(void* const* d_in, const int* in_sizes, int n_in,
                              void* d_out, int out_size, void* d_ws, size_t ws_size,
                              hipStream_t stream) {
    (void)in_sizes; (void)n_in; (void)out_size; (void)ws_size;
    const int N0 = 50000, N1 = 150000, N2 = 100000;
    const int EU0 = 300000, EU1 = 400000, ED1 = 400000, ED2 = 200000;
    const int Ns[3] = {N0, N1, N2};

    // params pytree flattened in insertion order -> leaves 0..183:
    //   per layer l (60 leaves), per dim d (20): up{W,b,g,be,mu,var}=+0..5,
    //   down{...}=+6..11, update{W1,b1,W2,b2,g,be,mu,var}=+12..19
    //   lin1_W=180, lin1_b=181, lin2_W=182, lin2_b=183
    auto P = [&](int i) { return (const float*)d_in[i]; };
    const float* x_in[3] = {(const float*)d_in[184], (const float*)d_in[185],
                            (const float*)d_in[186]};
    const int* up_index_0   = (const int*)d_in[187];
    const int* up_coface_0  = (const int*)d_in[188];
    const int* up_index_1   = (const int*)d_in[189];
    const int* up_coface_1  = (const int*)d_in[190];
    const int* down_index_1 = (const int*)d_in[191];
    const int* down_face_1  = (const int*)d_in[192];
    const int* down_index_2 = (const int*)d_in[193];
    const int* down_face_2  = (const int*)d_in[194];
    const int* batch[3] = {(const int*)d_in[195], (const int*)d_in[196],
                           (const int*)d_in[197]};

    float* ws = (float*)d_ws;
    size_t o = 0;   // in floats
    float *fA[3], *fB[3], *acc[3];
    for (int d = 0; d < 3; ++d) { fA[d] = ws + o; o += (size_t)Ns[d] * 64; }
    for (int d = 0; d < 3; ++d) { fB[d] = ws + o; o += (size_t)Ns[d] * 64; }
    for (int d = 0; d < 3; ++d) { acc[d] = ws + o; o += (size_t)Ns[d] * 64; }
    float* pooled = ws + o; o += 64 * 64;

    // ---- transpose + f16-convert all weights (one-time prologue) -----------
    auto tr = [&](const float* W, int K, int N, int Npad) {
        _Float16* Wt = (_Float16*)(ws + o); o += ((size_t)K * Npad + 1) / 2;
        transpose_w16_kernel<<<(K * Npad + 255) / 256, 256, 0, stream>>>(W, Wt, K, N, Npad);
        return Wt;
    };
    _Float16* msgWt[3][4];   // per layer: d0up, d1up, d1down, d2down
    _Float16* W1t[3][3];
    _Float16* W2t[3][3];
    for (int l = 0; l < 3; ++l) {
        const int ld = (l == 0) ? 32 : 64;
        const int base = l * 60;
        msgWt[l][0] = tr(P(base + 0),  2 * ld, ld, ld);   // d0.up.W
        msgWt[l][1] = tr(P(base + 20), 2 * ld, ld, ld);   // d1.up.W
        msgWt[l][2] = tr(P(base + 26), 2 * ld, ld, ld);   // d1.down.W
        msgWt[l][3] = tr(P(base + 46), 2 * ld, ld, ld);   // d2.down.W
        for (int d = 0; d < 3; ++d) {
            const int pb = base + d * 20;
            W1t[l][d] = tr(P(pb + 12), ld, 64, 64);       // update.W1
            W2t[l][d] = tr(P(pb + 14), 64, 64, 64);       // update.W2
        }
    }
    _Float16* headW1t = tr(P(180), 64, 64, 64);
    _Float16* headW2t = tr(P(182), 64, 10, 16);           // pad 10 -> 16 columns

    const float* cur[3] = {x_in[0], x_in[1], x_in[2]};
    float* nxt[3] = {fA[0], fA[1], fA[2]};
    float* alt[3] = {fB[0], fB[1], fB[2]};

    // persistent-grid sizing
    auto capped_grid = [](int items, int cap) {
        int b = (items + 127) / 128;               // 128 rows/edges per block
        return b < cap ? b : cap;
    };

    for (int l = 0; l < 3; ++l) {
        const int ld = (l == 0) ? 32 : 64;
        const int base = l * 60;
        // acc = 2*x
        for (int d = 0; d < 3; ++d) {
            size_t n = (size_t)Ns[d] * ld;
            scale2_kernel<<<(unsigned)((n + 255) / 256), 256, 0, stream>>>(cur[d], acc[d], n);
        }
        // messages: d0.up, d1.up, d1.down, d2.down
        struct MsgCfg {
            const float *x, *xo; const int *idx, *att; _Float16* wt;
            int bias_leaf; float* acc; int E;
        } cfgs[4] = {
            {cur[0], cur[1], up_index_0,   up_coface_0, msgWt[l][0], base + 1,  acc[0], EU0},
            {cur[1], cur[2], up_index_1,   up_coface_1, msgWt[l][1], base + 21, acc[1], EU1},
            {cur[1], cur[0], down_index_1, down_face_1, msgWt[l][2], base + 27, acc[1], ED1},
            {cur[2], cur[1], down_index_2, down_face_2, msgWt[l][3], base + 47, acc[2], ED2},
        };
        for (int c = 0; c < 4; ++c) {
            const MsgCfg& g = cfgs[c];
            const int blocks = capped_grid(g.E, 768);
            const int stride = blocks * 128;
            if (ld == 32)
                msg_kernel<32><<<blocks, 256, 0, stream>>>(g.x, g.xo,
                    g.idx, g.idx + g.E, g.att, g.wt,
                    P(g.bias_leaf), P(g.bias_leaf + 1), P(g.bias_leaf + 2),
                    P(g.bias_leaf + 3), P(g.bias_leaf + 4), g.acc, g.E, stride);
            else
                msg_kernel<64><<<blocks, 256, 0, stream>>>(g.x, g.xo,
                    g.idx, g.idx + g.E, g.att, g.wt,
                    P(g.bias_leaf), P(g.bias_leaf + 1), P(g.bias_leaf + 2),
                    P(g.bias_leaf + 3), P(g.bias_leaf + 4), g.acc, g.E, stride);
        }
        // updates: persistent, weights async-staged to LDS
        for (int d = 0; d < 3; ++d) {
            const int pb = base + d * 20;
            const int blocks = capped_grid(Ns[d], 512);
            const int stride = blocks * 128;
            if (ld == 32)
                update_kernel<32><<<blocks, 256, 0, stream>>>(acc[d],
                    W1t[l][d], P(pb+13), W2t[l][d], P(pb+15),
                    P(pb+16), P(pb+17), P(pb+18), P(pb+19), nxt[d], Ns[d], stride);
            else
                update_kernel<64><<<blocks, 256, 0, stream>>>(acc[d],
                    W1t[l][d], P(pb+13), W2t[l][d], P(pb+15),
                    P(pb+16), P(pb+17), P(pb+18), P(pb+19), nxt[d], Ns[d], stride);
        }
        for (int d = 0; d < 3; ++d) {
            cur[d] = nxt[d];
            float* t = nxt[d]; nxt[d] = alt[d]; alt[d] = t;
        }
    }

    // pooling + head
    zero_kernel<<<(64 * 64 + 255) / 256, 256, 0, stream>>>(pooled, 64 * 64);
    for (int d = 0; d < 3; ++d)
        pool_kernel<<<(Ns[d] * 64 + 255) / 256, 256, 0, stream>>>(cur[d], batch[d],
                                                                  pooled, Ns[d]);
    head_kernel<<<1, 128, 0, stream>>>(pooled, headW1t, P(181), headW2t, P(183),
                                       (float*)d_out);
}
